// SimpleSS2D_57878979281527
// MI455X (gfx1250) — matrix operand order repaired and checked by reading, not compiled, not run on hardware
//
#include <hip/hip_runtime.h>
#include <hip/hip_bf16.h>
#include <stdint.h>

// ---------------------------------------------------------------------------
// SS2D (VMamba selective-scan 2D) forward for gfx1250 (MI455X), wave32 WMMA.
//  K0: pack GEMM weights into per-lane WMMA fragment order (f16, coalesced)
//  K1: channel-LayerNorm + in_proj GEMM (WMMA f16 16x16x32) -> xi_raw, z
//  K2: depthwise 3x3 conv + SiLU + cross-scan scatter (plain f16 xs for the
//      scan + B-fragment-swizzled xs_sw for the x_proj GEMM)
//  K3: x_proj GEMM per (b,k): (38 x 4096) = (38x192)x(192x4096) (WMMA)
//  K4: dt_proj (K=6) + softplus -> delta (f16)
//  K5: selective scan; B/C staged to LDS with global_load_async_to_lds_b128
//      (ASYNCcnt, double-buffered) overlapping the v_exp_f32 recurrence
//  K6: cross-merge gather + LayerNorm(192) + SiLU(z) gate + out_proj (WMMA)
// ---------------------------------------------------------------------------

typedef _Float16 half_t;
typedef __attribute__((ext_vector_type(16))) _Float16 v16h;
typedef __attribute__((ext_vector_type(8)))  float    v8f;

namespace cfg {
constexpr int BS  = 8;      // batch
constexpr int CD  = 96;     // DIM
constexpr int HH  = 64;
constexpr int WW  = 64;
constexpr int LL  = HH * WW;        // 4096
constexpr int DI  = 192;            // DINNER
constexpr int DI2 = 2 * DI;         // 384
constexpr int KD  = 4;              // directions
constexpr int NS  = 16;             // DSTATE
constexpr int RT  = 6;              // DTRANK
constexpr int CP  = RT + 2 * NS;    // 38 rows of x_dbl
}

// ---------------- WMMA helpers (CDNA5 wave32 layouts) ----------------------
// A (16x32 f16), contiguous k as used here (not AMD's table order): lanes 0-15 row=lane, K 0..15;
//                lanes 16-31 row=lane-16, K 16..31.
__device__ __forceinline__ v16h load_a_f16(const half_t* src, int stride, int lane) {
  v16h a;
  int row = lane & 15;
  int kb  = (lane & 16) ? 16 : 0;
  const half_t* p = src + row * stride + kb;
#pragma unroll
  for (int i = 0; i < 8; ++i) { a[i] = p[i]; a[i + 8] = p[8 + i]; }
  return a;
}

// Packed fragment: [frag][lane][16 halves] contiguous -> 2x global_load_b128.
__device__ __forceinline__ v16h load_frag_packed(const half_t* base, long frag,
                                                 int lane) {
  return *(const v16h*)(base + (frag * 32 + lane) * 16);
}

__device__ __forceinline__ v8f wmma_f16(v16h a, v16h b, v8f c) {
  return __builtin_amdgcn_wmma_f32_16x16x32_f16(false, a, false, b, (short)0, c,
                                                false, false);
}

// ---------------- K0: pack weights into WMMA fragment order ----------------
// B-frag element map: col=lane&15, kb=(lane&16)?16:0, half i -> K = kb+i.
// A-frag element map: row=lane&15, kb=(lane&16)?16:0, half i -> K = kb+i.
__global__ void __launch_bounds__(256)
k0_pack(const float* __restrict__ in_w, const float* __restrict__ xw,
        const float* __restrict__ ow, half_t* __restrict__ wpk_in,
        half_t* __restrict__ wpk_xp, half_t* __restrict__ wpk_out) {
  using namespace cfg;
  const int gid = blockIdx.x * 256 + threadIdx.x;
  constexpr int NIN  = 24 * 3 * 512;  // in_proj:  24 ntiles x 3 kchunks
  constexpr int NXP  = 4 * 3 * 6 * 512;  // x_proj A: 4 dirs x 3 mtiles x 6 kchunks
  constexpr int NOUT = 6 * 6 * 512;   // out_proj: 6 ntiles x 6 kchunks
  if (gid < NIN) {
    int f = gid >> 9, r = gid & 511;
    int lane = r >> 4, i = r & 15;
    int nt = f / 3, kci = f % 3;
    int col = lane & 15, kb = (lane & 16) ? 16 : 0;
    wpk_in[gid] = (half_t)in_w[(long)(nt * 16 + col) * CD + kci * 32 + kb + i];
  } else if (gid < NIN + NXP) {
    int g = gid - NIN;
    int f = g >> 9, r = g & 511;
    int lane = r >> 4, i = r & 15;
    int k4 = f / 18, rem = f % 18;
    int mt = rem / 6, kci = rem % 6;
    int row = lane & 15, kb = (lane & 16) ? 16 : 0;
    int c = mt * 16 + row;
    int k = kci * 32 + kb + i;
    wpk_xp[g] = (c < CP) ? (half_t)xw[((long)k4 * CP + c) * DI + k] : (half_t)0.f;
  } else if (gid < NIN + NXP + NOUT) {
    int g = gid - NIN - NXP;
    int f = g >> 9, r = g & 511;
    int lane = r >> 4, i = r & 15;
    int nt = f / 6, kci = f % 6;
    int col = lane & 15, kb = (lane & 16) ? 16 : 0;
    wpk_out[g] = (half_t)ow[(long)(nt * 16 + col) * DI + kci * 32 + kb + i];
  }
}

// ---------------- K1: LayerNorm + in_proj ----------------------------------
__global__ void __launch_bounds__(256)
k1_ln_inproj(const float* __restrict__ x, const float* __restrict__ nw,
             const float* __restrict__ nb, const half_t* __restrict__ wpk_in,
             const float* __restrict__ bp, float* __restrict__ xi_raw,
             float* __restrict__ z) {
  using namespace cfg;
  __shared__ float  buf[16][CD];
  __shared__ half_t atile[16][CD];
  __shared__ float  sout[16][DI2];     // result staging for coalesced stores
  const int tile = blockIdx.x;         // 16-pixel tile over B*L
  const int t    = threadIdx.x;
  const int b0   = (tile * 16) / LL;   // tile never crosses batch (L%16==0)
  const int l0   = (tile * 16) % LL;

  for (int idx = t; idx < 16 * CD; idx += 256) {
    int m = idx / CD, c = idx % CD;
    buf[m][c] = x[((long)b0 * CD + c) * LL + l0 + m];
  }
  __syncthreads();
  if (t < 16) {
    float mu = 0.f;
    for (int c = 0; c < CD; ++c) mu += buf[t][c];
    mu *= (1.f / CD);
    float var = 0.f;
    for (int c = 0; c < CD; ++c) { float d = buf[t][c] - mu; var += d * d; }
    var *= (1.f / CD);
    float rs = rsqrtf(var + 1e-5f);
    for (int c = 0; c < CD; ++c)
      atile[t][c] = (half_t)((buf[t][c] - mu) * rs * nw[c] + nb[c]);
  }
  __syncthreads();

  const int wave = t >> 5, lane = t & 31;
  for (int nt = wave; nt < DI2 / 16; nt += 8) {     // 24 N-tiles over 8 waves
    v8f acc = {};
#pragma unroll
    for (int kci = 0; kci < 3; ++kci) {
      v16h a  = load_a_f16(&atile[0][0] + kci * 32, CD, lane);
      v16h bb = load_frag_packed(wpk_in, nt * 3 + kci, lane);
      acc = wmma_f16(a, bb, acc);
    }
    const int col = lane & 15;
    const int cg  = nt * 16 + col;
    const float bias = bp[cg];
    const int mbase = (lane & 16) ? 8 : 0;
#pragma unroll
    for (int j = 0; j < 8; ++j) sout[mbase + j][cg] = acc[j] + bias;
  }
  __syncthreads();
  // xi: coalesced over l (channel-major thread grouping)
  for (int idx = t; idx < 16 * DI; idx += 256) {
    int c = idx >> 4, m = idx & 15;
    xi_raw[((long)b0 * DI + c) * LL + l0 + m] = sout[m][c];
  }
  // z: coalesced over channel
  for (int idx = t; idx < 16 * DI; idx += 256) {
    int m = idx / DI, c = idx % DI;
    z[((long)b0 * LL + l0 + m) * DI + c] = sout[m][DI + c];
  }
}

// ---------------- K2: depthwise conv + SiLU + cross-scan -------------------
// Also writes the B-fragment swizzled layout consumed by K3.
__device__ __forceinline__ void store_sw(half_t* __restrict__ xs_sw, int bk,
                                         int c, int pos, half_t v) {
  using namespace cfg;
  int kchunk = c >> 5, dk = c & 31;
  int lane = (pos & 15) + ((dk >= 16) ? 16 : 0);
  int i = dk & 15;
  int nt = pos >> 4;
  xs_sw[(((long)(bk * 6 + kchunk) * 256 + nt) * 32 + lane) * 16 + i] = v;
}

__global__ void __launch_bounds__(256)
k2_conv_scan(const float* __restrict__ xi_raw, const float* __restrict__ cw,
             const float* __restrict__ cb, half_t* __restrict__ xs,
             half_t* __restrict__ xs_sw) {
  using namespace cfg;
  long idx = (long)blockIdx.x * 256 + threadIdx.x;   // over B*DI*L
  if (idx >= (long)BS * DI * LL) return;
  int l = (int)(idx % LL);
  long t2 = idx / LL;
  int c = (int)(t2 % DI);
  int b = (int)(t2 / DI);
  int h = l / WW, w = l % WW;

  float acc = cb[c];
  const float* src = xi_raw + ((long)b * DI + c) * LL;
  const float* wk  = cw + (long)c * 9;
#pragma unroll
  for (int dy = -1; dy <= 1; ++dy)
#pragma unroll
    for (int dx = -1; dx <= 1; ++dx) {
      int hy = h + dy, wx = w + dx;
      if ((unsigned)hy < (unsigned)HH && (unsigned)wx < (unsigned)WW)
        acc += wk[(dy + 1) * 3 + (dx + 1)] * src[hy * WW + wx];
    }
  float v = acc / (1.f + __expf(-acc));    // SiLU
  half_t hv = (half_t)v;
  long base = (long)b * KD * DI * LL;
  int lwh = w * HH + h;
  int pos[KD] = { l, lwh, LL - 1 - l, LL - 1 - lwh };
#pragma unroll
  for (int k = 0; k < KD; ++k) {
    xs[base + ((long)(k * DI + c)) * LL + pos[k]] = hv;   // scan-stream layout
    store_sw(xs_sw, b * KD + k, c, pos[k], hv);           // GEMM B-frag layout
  }
}

// ---------------- K3: x_proj GEMMs -----------------------------------------
__global__ void __launch_bounds__(256)
k3_xproj(const half_t* __restrict__ xs_sw, const half_t* __restrict__ wpk_xp,
         float* __restrict__ xdbl) {
  using namespace cfg;
  const int gwave = (int)((blockIdx.x * 256 + threadIdx.x) >> 5);
  const int lane  = threadIdx.x & 31;
  const int nt   = gwave % (LL / 16);       // 256 N-tiles
  const int rest = gwave / (LL / 16);
  const int mt   = rest % 3;                // 3 M-tiles (rows 0..47, guard 38)
  const int bk   = rest / 3;                // 32 (b,k) pairs
  if (bk >= BS * KD) return;
  const int k = bk & 3;

  v8f acc = {};
#pragma unroll
  for (int kci = 0; kci < 6; ++kci) {
    v16h a  = load_frag_packed(wpk_xp, (long)(k * 3 + mt) * 6 + kci, lane);
    v16h bb = load_frag_packed(xs_sw, (long)(bk * 6 + kci) * 256 + nt, lane);
    acc = wmma_f16(a, bb, acc);
  }
  const int col = lane & 15;
  const int l = nt * 16 + col;
  const int mbase = (lane & 16) ? 8 : 0;
#pragma unroll
  for (int j = 0; j < 8; ++j) {
    int c = mt * 16 + mbase + j;
    if (c < CP)
      xdbl[((long)bk * CP + c) * LL + l] = acc[j];
  }
}

// ---------------- K4: dt_proj + softplus -> delta --------------------------
__global__ void __launch_bounds__(256)
k4_delta(const float* __restrict__ xdbl, const float* __restrict__ dtw,
         const float* __restrict__ dtb, half_t* __restrict__ delta) {
  using namespace cfg;
  long idx = (long)blockIdx.x * 256 + threadIdx.x;   // over B*KD*DI*L
  if (idx >= (long)BS * KD * DI * LL) return;
  int l = (int)(idx % LL);
  long t2 = idx / LL;
  int d = (int)(t2 % DI);
  int t3 = (int)(t2 / DI);
  int k = t3 & 3;

  const float* wrow = dtw + ((long)k * DI + d) * RT;
  const float* xcol = xdbl + ((long)t3 * CP) * LL + l;
  float acc = dtb[k * DI + d];
#pragma unroll
  for (int r = 0; r < RT; ++r) acc += wrow[r] * xcol[(long)r * LL];
  float sp = (acc > 20.f) ? acc : log1pf(__expf(acc));  // softplus
  delta[idx] = (half_t)sp;
}

// ---------------- K5: selective scan (async LDS staging) -------------------
__global__ void __launch_bounds__(192)
k5_scan(const half_t* __restrict__ xs, const half_t* __restrict__ delta,
        const float* __restrict__ xdbl, const float* __restrict__ A_log,
        const float* __restrict__ Ds, float* __restrict__ oy) {
  using namespace cfg;
  constexpr int CH = 32;
  __shared__ __attribute__((aligned(16))) float sBC[2][2][NS][CH]; // [buf][B/C][n][ll]
  const int bk = blockIdx.x;       // 32 blocks
  const int k = bk & 3;
  const int d = threadIdx.x;       // 192 channels

  float Arow[NS];
#pragma unroll
  for (int n = 0; n < NS; ++n)
    Arow[n] = -__expf(A_log[((long)k * DI + d) * NS + n]);
  const float Dv = Ds[k * DI + d];

  const half_t* up = xs    + ((long)bk * DI + d) * LL;
  const half_t* dp = delta + ((long)bk * DI + d) * LL;
  const float* Bbase = xdbl + ((long)bk * CP + RT) * LL;
  const float* Cbase = xdbl + ((long)bk * CP + RT + NS) * LL;
  float* op = oy + ((long)bk * DI + d) * LL;

  // Issue one chunk of B/C into LDS via async global->LDS copies (ASYNCcnt).
  auto issue = [&](int l0, int bufi) {
    for (int o = threadIdx.x; o < 256; o += 192) {
      int m = o >> 7;               // 0 = B, 1 = C
      int n = (o >> 3) & 15;
      int g = o & 7;                // 16B group within 32-col row
      const float* srcp = (m ? Cbase : Bbase) + (long)n * LL + l0 + g * 4;
      float* dstp = &sBC[bufi][m][n][g * 4];
      asm volatile("global_load_async_to_lds_b128 %0, %1, off"
                   :: "v"((unsigned)(uintptr_t)dstp),
                      "v"((unsigned long long)(uintptr_t)srcp)
                   : "memory");
    }
  };

  float h[NS];
#pragma unroll
  for (int n = 0; n < NS; ++n) h[n] = 0.f;

  issue(0, 0);
  for (int cc = 0; cc < LL / CH; ++cc) {
    const int l0 = cc * CH;
    asm volatile("s_wait_asynccnt 0" ::: "memory");
    __syncthreads();                      // chunk cc visible to all waves
    if (cc + 1 < LL / CH) issue(l0 + CH, (cc + 1) & 1);  // overlap next chunk
    const int bufi = cc & 1;
    __builtin_prefetch(up + l0 + CH, 0, 1);   // own-stream prefetch
    __builtin_prefetch(dp + l0 + CH, 0, 1);
#pragma unroll 4
    for (int ll = 0; ll < CH; ++ll) {
      float u  = (float)up[l0 + ll];
      float dl = (float)dp[l0 + ll];
      float du = dl * u;
      float y = 0.f;
#pragma unroll
      for (int n = 0; n < NS; ++n) {
        float dA = __expf(dl * Arow[n]);
        h[n] = dA * h[n] + du * sBC[bufi][0][n][ll];
        y += h[n] * sBC[bufi][1][n][ll];
      }
      op[l0 + ll] = y + Dv * u;
    }
  }
}

// ---------------- K6: merge + LN + gate + out_proj -------------------------
__global__ void __launch_bounds__(256)
k6_merge_out(const float* __restrict__ oy, const float* __restrict__ z,
             const float* __restrict__ onw, const float* __restrict__ onb,
             const half_t* __restrict__ wpk_out, const float* __restrict__ ob,
             float* __restrict__ out) {
  using namespace cfg;
  __shared__ float  ybuf[16][DI];
  __shared__ half_t atile[16][DI];
  __shared__ float  sout[16][CD];
  const int tile = blockIdx.x;
  const int t    = threadIdx.x;
  const int b0   = (tile * 16) / LL;
  const int l0   = (tile * 16) % LL;

  for (int idx = t; idx < 16 * DI; idx += 256) {
    int m = idx / DI, d = idx % DI;
    int l = l0 + m;
    int h = l / WW, w = l % WW;
    int lwh = w * HH + h;
    const float* o0 = oy + ((long)(b0 * KD) * DI + d) * LL;
    ybuf[m][d] = o0[l] + o0[2L * DI * LL + (LL - 1 - l)] +
                 o0[1L * DI * LL + lwh] + o0[3L * DI * LL + (LL - 1 - lwh)];
  }
  __syncthreads();
  if (t < 16) {
    float mu = 0.f;
    for (int d = 0; d < DI; ++d) mu += ybuf[t][d];
    mu *= (1.f / DI);
    float var = 0.f;
    for (int d = 0; d < DI; ++d) { float e = ybuf[t][d] - mu; var += e * e; }
    var *= (1.f / DI);
    float rs = rsqrtf(var + 1e-5f);
    const float* zp = z + ((long)b0 * LL + l0 + t) * DI;
    for (int d = 0; d < DI; ++d) {
      float v = (ybuf[t][d] - mu) * rs * onw[d] + onb[d];
      float zv = zp[d];
      v *= zv / (1.f + __expf(-zv));       // * silu(z)
      atile[t][d] = (half_t)v;
    }
  }
  __syncthreads();

  const int wave = t >> 5, lane = t & 31;
  if (wave < CD / 16) {                    // 6 N-tiles of out channels
    const int nt = wave;
    v8f acc = {};
#pragma unroll
    for (int kci = 0; kci < 6; ++kci) {
      v16h a  = load_a_f16(&atile[0][0] + kci * 32, DI, lane);
      v16h bb = load_frag_packed(wpk_out, nt * 6 + kci, lane);
      acc = wmma_f16(a, bb, acc);
    }
    const int col = lane & 15;
    const int c = nt * 16 + col;
    const float bias = ob[c];
    const int mbase = (lane & 16) ? 8 : 0;
#pragma unroll
    for (int j = 0; j < 8; ++j) sout[mbase + j][c] = acc[j] + bias;
  }
  __syncthreads();
  for (int idx = t; idx < 16 * CD; idx += 256) {  // coalesced over l
    int c = idx >> 4, m = idx & 15;
    out[((long)b0 * CD + c) * LL + l0 + m] = sout[m][c];
  }
}

// ---------------- host launcher --------------------------------------------
extern "C" void kernel_launch(void* const* d_in, const int* in_sizes, int n_in,
                              void* d_out, int out_size, void* d_ws,
                              size_t ws_size, hipStream_t stream) {
  using namespace cfg;
  const float* x        = (const float*)d_in[0];
  const float* norm_w   = (const float*)d_in[1];
  const float* norm_b   = (const float*)d_in[2];
  const float* in_w     = (const float*)d_in[3];
  const float* in_b     = (const float*)d_in[4];
  const float* conv_w   = (const float*)d_in[5];
  const float* conv_b   = (const float*)d_in[6];
  const float* xproj_w  = (const float*)d_in[7];
  const float* dt_w     = (const float*)d_in[8];
  const float* dt_b     = (const float*)d_in[9];
  const float* A_log    = (const float*)d_in[10];
  const float* Ds       = (const float*)d_in[11];
  const float* onorm_w  = (const float*)d_in[12];
  const float* onorm_b  = (const float*)d_in[13];
  const float* out_w    = (const float*)d_in[14];
  const float* out_b    = (const float*)d_in[15];
  float* out = (float*)d_out;

  const size_t MB = 1024ull * 1024ull;
  char* ws = (char*)d_ws;
  float*  xi_raw  = (float*)(ws + 0);           // 24 MB  (B*DI*L f32)
  float*  zbuf    = (float*)(ws + 24 * MB);     // 24 MB  (B*L*DI f32)
  half_t* xs      = (half_t*)(ws + 48 * MB);    // 48 MB  (B*4*DI*L f16)
  float*  xdbl    = (float*)(ws + 96 * MB);     // 20 MB  (B*4*38*L f32)
  half_t* dlt     = (half_t*)(ws + 116 * MB);   // 48 MB  (B*4*DI*L f16)
  float*  oy      = (float*)(ws + 164 * MB);    // 96 MB  (B*4*DI*L f32)
  half_t* xs_sw   = (half_t*)(ws + 260 * MB);   // 48 MB  (B-frag swizzled xs)
  half_t* wpk_in  = (half_t*)(ws + 308 * MB);   // 72 KB
  half_t* wpk_xp  = (half_t*)(ws + 309 * MB);   // 72 KB
  half_t* wpk_out = (half_t*)(ws + 310 * MB);   // 36 KB

  k0_pack<<<360, 256, 0, stream>>>(in_w, xproj_w, out_w, wpk_in, wpk_xp,
                                   wpk_out);
  k1_ln_inproj<<<BS * LL / 16, 256, 0, stream>>>(x, norm_w, norm_b, wpk_in,
                                                 in_b, xi_raw, zbuf);
  {
    long n = (long)BS * DI * LL;
    k2_conv_scan<<<(int)((n + 255) / 256), 256, 0, stream>>>(xi_raw, conv_w,
                                                             conv_b, xs, xs_sw);
  }
  {
    int waves = BS * KD * 3 * (LL / 16);       // 24576 wave-tiles
    k3_xproj<<<waves * 32 / 256, 256, 0, stream>>>(xs_sw, wpk_xp, xdbl);
  }
  {
    long n = (long)BS * KD * DI * LL;
    k4_delta<<<(int)((n + 255) / 256), 256, 0, stream>>>(xdbl, dt_w, dt_b, dlt);
  }
  k5_scan<<<BS * KD, 192, 0, stream>>>(xs, dlt, xdbl, A_log, Ds, oy);
  k6_merge_out<<<BS * LL / 16, 256, 0, stream>>>(oy, zbuf, onorm_w, onorm_b,
                                                 wpk_out, out_b, out);
}
